// SwinPoseNet_25151328485420
// MI455X (gfx1250) — matrix operand order repaired and checked by reading, not compiled, not run on hardware
//
#include <hip/hip_runtime.h>
#include <hip/hip_bf16.h>
#include <math.h>

// ---------------------------------------------------------------------------
// Swin stage for MI455X (gfx1250, wave32, WMMA 16x16x32 f16 -> f32)
// ---------------------------------------------------------------------------

typedef __attribute__((ext_vector_type(16))) _Float16 v16h;
typedef __attribute__((ext_vector_type(8)))  _Float16 v8h;
typedef __attribute__((ext_vector_type(8)))  float    v8f;

#define WMMA_F16(a, b, c) \
  __builtin_amdgcn_wmma_f32_16x16x32_f16(false, (a), false, (b), (short)0, (c), false, false)

// Problem constants
static constexpr int    HH   = 224;              // spatial after patch merge
static constexpr size_t MROW = 2ull * 224 * 224; // 100352 token rows
static constexpr int    DIMC = 192;
static constexpr int    QKVN = 576;
static constexpr int    MLPN = 768;
static constexpr int    PMK  = 384;
static constexpr int    NWW  = 32;               // windows per axis (224/7)

enum { EPI_NONE = 0, EPI_BIAS = 1, EPI_BIAS_RES = 2, EPI_BIAS_GELU = 3 };

// ---------------- WMMA fragment loaders (from LDS, padded strides) ----------
// A 16x32 f16: lanes 0-15 & 16-31 both hold rows M=0..15; lanes<16 take K
// {0..7,16..23}, lanes>=16 take K {8..15,24..31}. ldk must be a multiple of 8.
static __device__ inline v16h load_a_frag(const _Float16* S, int rowBase, int lane,
                                          int ldk, int k0) {
  int r    = rowBase + (lane & 15);
  int koff = k0 + ((lane & 16) >> 1);            // +8 for upper half-wave
  const _Float16* p = S + (size_t)r * ldk + koff;
  v8h lo = *(const v8h*)p;
  v8h hi = *(const v8h*)(p + 16);
  return __builtin_shufflevector(lo, hi, 0,1,2,3,4,5,6,7,8,9,10,11,12,13,14,15);
}
// B 32x16 f16 from K-contiguous storage: lane column = tile + lane%16,
// lanes 0-15 hold K {0..7,16..23}, lanes 16-31 hold K {8..15,24..31}.
// Two v8h loads so padded (non-32B-aligned) strides are legal.
static __device__ inline v16h load_b_frag(const _Float16* S, int colBase, int lane,
                                          int ldk, int k0) {
  int c    = colBase + (lane & 15);
  int koff = k0 + ((lane & 16) >> 1);            // +8 for upper half-wave
  const _Float16* p = S + (size_t)c * ldk + koff;
  v8h lo = *(const v8h*)p;
  v8h hi = *(const v8h*)(p + 16);
  return __builtin_shufflevector(lo, hi, 0,1,2,3,4,5,6,7,8,9,10,11,12,13,14,15);
}

// ---------------- Generic WMMA GEMM: C[M,N] = A[M,K] * Bt[N,K]^T ------------
// 128x64 block tile, BK=32, 128 threads = 4 waves (2x2), wave tile 64x32.
// Register-staged prefetch of the next k-tile overlaps HBM latency with WMMAs.
template <int EPI>
__global__ __launch_bounds__(128) void gemm_f16_kernel(
    const _Float16* __restrict__ A,   // [M][K] f16
    const _Float16* __restrict__ Bt,  // [N][K] f16 (pre-transposed weight)
    const float* __restrict__ bias,   // [N] or null
    const float* __restrict__ res,    // [M][N] residual (EPI_BIAS_RES)
    float* __restrict__ Cf,           // f32 out or null
    _Float16* __restrict__ Ch,        // f16 out or null
    int Ndim, int K) {
  constexpr int LDA = 40;  // padded stride (halves): r*20 mod 64 -> no bank conflicts
  __shared__ __align__(32) _Float16 As[128 * LDA];
  __shared__ __align__(32) _Float16 Bs[64 * LDA];

  const int tid  = threadIdx.x;
  const int lane = tid & 31;
  const int wave = tid >> 5;
  const int bm   = blockIdx.y * 128;
  const int bn   = blockIdx.x * 64;
  const int wm   = (wave >> 1) * 64;
  const int wn   = (wave & 1) * 32;

  v8f acc[4][2] = {};
  v8h ra[4], rb[2];

  auto gload = [&](int k0) {
#pragma unroll
    for (int j = 0; j < 4; ++j) {   // 128x32 A tile: 512 8-half chunks
      int i = tid + j * 128, r = i >> 2, c = (i & 3) << 3;
      ra[j] = *(const v8h*)&A[(size_t)(bm + r) * K + k0 + c];
    }
#pragma unroll
    for (int j = 0; j < 2; ++j) {   // 64x32 B tile: 256 chunks
      int i = tid + j * 128, r = i >> 2, c = (i & 3) << 3;
      rb[j] = *(const v8h*)&Bt[(size_t)(bn + r) * K + k0 + c];
    }
  };

  gload(0);
  for (int k0 = 0; k0 < K; k0 += 32) {
    __syncthreads();  // previous iteration done reading LDS
#pragma unroll
    for (int j = 0; j < 4; ++j) {
      int i = tid + j * 128, r = i >> 2, c = (i & 3) << 3;
      *(v8h*)&As[r * LDA + c] = ra[j];
    }
#pragma unroll
    for (int j = 0; j < 2; ++j) {
      int i = tid + j * 128, r = i >> 2, c = (i & 3) << 3;
      *(v8h*)&Bs[r * LDA + c] = rb[j];
    }
    __syncthreads();
    if (k0 + 32 < K) gload(k0 + 32);  // prefetch next tile while computing

    v16h af[4], bf[2];
#pragma unroll
    for (int mt = 0; mt < 4; ++mt) af[mt] = load_a_frag(As, wm + mt * 16, lane, LDA, 0);
#pragma unroll
    for (int nt = 0; nt < 2; ++nt) bf[nt] = load_b_frag(Bs, wn + nt * 16, lane, LDA, 0);
#pragma unroll
    for (int mt = 0; mt < 4; ++mt)
#pragma unroll
      for (int nt = 0; nt < 2; ++nt) acc[mt][nt] = WMMA_F16(af[mt], bf[nt], acc[mt][nt]);
  }

  // Epilogue. C layout: VGPR i -> row (i + 8*(lane>=16)), col = lane%16.
  const int lcol = lane & 15;
  const int rhi  = (lane & 16) >> 1;
#pragma unroll
  for (int mt = 0; mt < 4; ++mt) {
#pragma unroll
    for (int nt = 0; nt < 2; ++nt) {
      int   col = bn + wn + nt * 16 + lcol;
      float bv  = (EPI != EPI_NONE) ? bias[col] : 0.0f;
#pragma unroll
      for (int i = 0; i < 8; ++i) {
        int    row = bm + wm + mt * 16 + rhi + i;
        size_t idx = (size_t)row * Ndim + col;
        float  v   = acc[mt][nt][i] + bv;
        if (EPI == EPI_BIAS_RES) v += res[idx];
        if (EPI == EPI_BIAS_GELU) v = 0.5f * v * (1.0f + erff(v * 0.70710678118654752f));
        if (Cf) Cf[idx] = v;
        if (Ch) Ch[idx] = (_Float16)v;
      }
    }
  }
}

// ---------------- Windowed attention: one (batch, window, head) per block ---
__global__ __launch_bounds__(128) void attn_kernel(
    const _Float16* __restrict__ qkv,  // [M][576] f16, layout (3, head, hd)
    const float* __restrict__ pos,     // [13][13]
    _Float16* __restrict__ out,        // [M][192] f16
    int shifted) {
  constexpr int LQK = 40;  // q/k row stride (halves), conflict-free frag reads
  constexpr int LVP = 72;  // vt/Ps stride (halves)
  constexpr int LSS = 65;  // Ss stride (f32): bank = (r+j) mod 64
  __shared__ __align__(32) _Float16 qs[64 * LQK];
  __shared__ __align__(32) _Float16 ks[64 * LQK];
  __shared__ __align__(32) _Float16 vt[32 * LVP];  // V transposed: [d][j]
  __shared__ __align__(32) float    Ss[64 * LSS];
  __shared__ __align__(32) _Float16 Ps[64 * LVP];
  __shared__ int   rowmap[64];
  __shared__ float posl[169];

  const int tid  = threadIdx.x;
  const int lane = tid & 31;
  const int wave = tid >> 5;

  const int bid  = blockIdx.x;            // ((b*1024 + win) * 6 + head)
  const int head = bid % 6;
  const int win  = (bid / 6) % (NWW * NWW);
  const int b    = bid / (6 * NWW * NWW);
  const int wi   = win / NWW, wj = win % NWW;
  const int sh   = shifted ? 3 : 0;

  for (int r = tid; r < 64; r += 128) {
    int m = -1;
    if (r < 49) {
      int pi = r / 7, pj = r % 7;
      int gy = wi * 7 + pi + sh; if (gy >= HH) gy -= HH;
      int gx = wj * 7 + pj + sh; if (gx >= HH) gx -= HH;
      m = (b * HH + gy) * HH + gx;
    }
    rowmap[r] = m;
  }
  for (int i = tid; i < 169; i += 128) posl[i] = pos[i];
  __syncthreads();

  // Load Q, K rows and V transposed (rows >= 49 zero-padded).
#pragma unroll
  for (int i = tid; i < 256; i += 128) {
    int r = i >> 2, c = (i & 3) << 3;
    int m = rowmap[r];
    v8h qv = {}, kv = {}, vv = {};
    if (m >= 0) {
      const _Float16* p = qkv + (size_t)m * QKVN + head * 32 + c;
      qv = *(const v8h*)p;
      kv = *(const v8h*)(p + 192);
      vv = *(const v8h*)(p + 384);
    }
    *(v8h*)&qs[r * LQK + c] = qv;
    *(v8h*)&ks[r * LQK + c] = kv;
#pragma unroll
    for (int u = 0; u < 8; ++u) vt[(c + u) * LVP + r] = vv[u];
  }
  __syncthreads();

  // S = Q * K^T : wave owns one 16-row tile, 4 column tiles, K=32 (one step).
  {
    v16h a = load_a_frag(qs, wave * 16, lane, LQK, 0);
    const int lcol = lane & 15, rhi = (lane & 16) >> 1;
#pragma unroll
    for (int nt = 0; nt < 4; ++nt) {
      v16h bf = load_b_frag(ks, nt * 16, lane, LQK, 0);
      v8f  c  = {};
      c = WMMA_F16(a, bf, c);
#pragma unroll
      for (int u = 0; u < 8; ++u)
        Ss[(wave * 16 + rhi + u) * LSS + nt * 16 + lcol] = c[u];
    }
  }
  __syncthreads();

  // Scale + rel-pos bias + shift masks + softmax; one thread per row.
  if (tid < 64) {
    const int  r  = tid;
    const int  pi = r / 7, pj = r % 7;
    const bool rv = (r < 49);
    float mx = -3.0e38f;
    for (int j = 0; j < 64; ++j) {
      float v = -3.0e38f;
      if (rv && j < 49) {
        int   ci = j / 7, cj = j % 7;
        float s  = Ss[r * LSS + j] * 0.17677669529663689f;     // 32^-0.5
        s += posl[(ci - pi + 6) * 13 + (cj - pj + 6)];
        bool masked = false;
        if (shifted) {
          if (wi == NWW - 1 && ((pi >= 4) != (ci >= 4))) masked = true;
          if (wj == NWW - 1 && ((pj >= 4) != (cj >= 4))) masked = true;
        }
        if (!masked) v = s;
      }
      Ss[r * LSS + j] = v;
      mx = fmaxf(mx, v);
    }
    float sum = 0.0f;
    for (int j = 0; j < 64; ++j) {
      float e = __expf(Ss[r * LSS + j] - mx);
      Ss[r * LSS + j] = e;
      sum += e;
    }
    float inv = 1.0f / sum;
    for (int j = 0; j < 64; ++j) Ps[r * LVP + j] = (_Float16)(Ss[r * LSS + j] * inv);
  }
  __syncthreads();

  // O = P * V : K=64 (two steps), 2 column tiles of 16 (HD=32).
  {
    const int lcol = lane & 15, rhi = (lane & 16) >> 1;
#pragma unroll
    for (int nt = 0; nt < 2; ++nt) {
      v8f c = {};
#pragma unroll
      for (int kk = 0; kk < 2; ++kk) {
        v16h a  = load_a_frag(Ps, wave * 16, lane, LVP, kk * 32);
        v16h bf = load_b_frag(vt, nt * 16, lane, LVP, kk * 32);
        c = WMMA_F16(a, bf, c);
      }
#pragma unroll
      for (int u = 0; u < 8; ++u) {
        int r = wave * 16 + rhi + u;
        int m = rowmap[r];
        if (m >= 0)
          out[(size_t)m * DIMC + head * 32 + nt * 16 + lcol] = (_Float16)c[u];
      }
    }
  }
}

// ---------------- LayerNorm (f32 in -> f16 out), one wave per 192-elem row --
__global__ __launch_bounds__(256) void layernorm_kernel(
    const float* __restrict__ x, const float* __restrict__ g,
    const float* __restrict__ be, _Float16* __restrict__ y) {
  const int row  = blockIdx.x * 8 + (threadIdx.x >> 5);
  const int lane = threadIdx.x & 31;
  const float* xr = x + (size_t)row * DIMC;
  float v[6], s = 0.0f;
#pragma unroll
  for (int i = 0; i < 6; ++i) { v[i] = xr[lane + 32 * i]; s += v[i]; }
#pragma unroll
  for (int o = 16; o > 0; o >>= 1) s += __shfl_xor(s, o, 32);
  const float mean = s * (1.0f / 192.0f);
  float var = 0.0f;
#pragma unroll
  for (int i = 0; i < 6; ++i) { float d = v[i] - mean; var += d * d; }
#pragma unroll
  for (int o = 16; o > 0; o >>= 1) var += __shfl_xor(var, o, 32);
  const float rs = rsqrtf(var * (1.0f / 192.0f) + 1e-5f);
  _Float16* yr = y + (size_t)row * DIMC;
#pragma unroll
  for (int i = 0; i < 6; ++i) {
    int c = lane + 32 * i;
    yr[c] = (_Float16)((v[i] - mean) * rs * g[c] + be[c]);
  }
}

// ---------------- PatchMerging gather: x[B,96,448,448] -> A[M][384] f16 -----
__global__ __launch_bounds__(256) void patch_gather_kernel(
    const float* __restrict__ x, _Float16* __restrict__ A) {
  size_t idx = (size_t)blockIdx.x * 256 + threadIdx.x;
  if (idx >= MROW * PMK) return;
  int    f = (int)(idx % PMK);
  size_t m = idx / PMK;
  int j = (int)(m % HH);
  int i = (int)((m / HH) % HH);
  int b = (int)(m / ((size_t)HH * HH));
  int c = f >> 2, p1 = (f >> 1) & 1, p2 = f & 1;
  A[idx] = (_Float16)x[(((size_t)b * 96 + c) * 448 + (2 * i + p1)) * 448 + (2 * j + p2)];
}

// ---------------- Weight cast + transpose: W[K][N] f32 -> Wt[N][K] f16 ------
__global__ __launch_bounds__(256) void wcast_t_kernel(
    const float* __restrict__ W, _Float16* __restrict__ Wt, int K, int N) {
  int idx = blockIdx.x * 256 + threadIdx.x;
  if (idx >= K * N) return;
  int n = idx / K, k = idx % K;
  Wt[idx] = (_Float16)W[(size_t)k * N + n];
}

// ---------------- Final NHWC(f32) -> NCHW(f32) ------------------------------
__global__ __launch_bounds__(256) void to_nchw_kernel(
    const float* __restrict__ t, float* __restrict__ o) {
  size_t idx = (size_t)blockIdx.x * 256 + threadIdx.x;
  if (idx >= MROW * DIMC) return;
  int xw = (int)(idx % HH);
  int yh = (int)((idx / HH) % HH);
  int c  = (int)((idx / ((size_t)HH * HH)) % DIMC);
  int b  = (int)(idx / ((size_t)HH * HH * DIMC));
  o[idx] = t[(((size_t)b * HH + yh) * HH + xw) * DIMC + c];
}

// ---------------------------------------------------------------------------
extern "C" void kernel_launch(void* const* d_in, const int* in_sizes, int n_in,
                              void* d_out, int out_size, void* d_ws, size_t ws_size,
                              hipStream_t stream) {
  (void)in_sizes; (void)n_in; (void)out_size; (void)ws_size;

  const float* x    = (const float*)d_in[0];
  const float* pm_w = (const float*)d_in[1];
  const float* pm_b = (const float*)d_in[2];
  const float* P[27];
  for (int i = 0; i < 27; ++i) P[i] = (const float*)d_in[i];

  char* ws = (char*)d_ws;
  float*    t      = (float*)(ws + 0);                         // residual f32 [M][192]
  _Float16* ln16   = (_Float16*)(ws + 77070336ull);            // [M][192] f16
  _Float16* qkv16  = (_Float16*)(ws + 115605504ull);           // [M][576] f16
  _Float16* attn16 = (_Float16*)(ws + 231211008ull);           // [M][192] f16
  _Float16* h16    = (_Float16*)(ws + 269746176ull);           // [M][768] f16
  _Float16* Apm    = h16;                                      // alias: [M][384] f16
  _Float16* wbase  = (_Float16*)(ws + 423886848ull);

  _Float16* pmT    = wbase;                    // 192*384
  _Float16* qkvT_r = pmT + 73728;              // 576*192
  _Float16* woT_r  = qkvT_r + 110592;          // 192*192
  _Float16* w1T_r  = woT_r + 36864;            // 768*192
  _Float16* w2T_r  = w1T_r + 147456;           // 192*768
  _Float16* qkvT_s = w2T_r + 147456;
  _Float16* woT_s  = qkvT_s + 110592;
  _Float16* w1T_s  = woT_s + 36864;
  _Float16* w2T_s  = w1T_s + 147456;

  const int MB = (int)(MROW / 128);  // 784 row-tiles of 128

  auto wc = [&](const float* W, _Float16* Wt, int K, int N) {
    wcast_t_kernel<<<(K * N + 255) / 256, 256, 0, stream>>>(W, Wt, K, N);
  };
  wc(pm_w, pmT, PMK, DIMC);
  wc(P[5],  qkvT_r, DIMC, QKVN);  wc(P[7],  woT_r, DIMC, DIMC);
  wc(P[11], w1T_r,  DIMC, MLPN);  wc(P[13], w2T_r, MLPN, DIMC);
  wc(P[17], qkvT_s, DIMC, QKVN);  wc(P[19], woT_s, DIMC, DIMC);
  wc(P[23], w1T_s,  DIMC, MLPN);  wc(P[25], w2T_s, MLPN, DIMC);

  // Patch merge: gather + GEMM (M,384)x(384,192) + bias -> t (f32)
  patch_gather_kernel<<<(unsigned)((MROW * PMK + 255) / 256), 256, 0, stream>>>(x, Apm);
  gemm_f16_kernel<EPI_BIAS><<<dim3(DIMC / 64, MB), 128, 0, stream>>>(
      Apm, pmT, pm_b, nullptr, t, nullptr, DIMC, PMK);

  auto run_block = [&](const float* ln1g, const float* ln1b, const _Float16* qkvT,
                       const float* pos, const _Float16* woT, const float* bo,
                       const float* ln2g, const float* ln2b, const _Float16* w1T,
                       const float* b1, const _Float16* w2T, const float* b2,
                       int shifted) {
    layernorm_kernel<<<(unsigned)(MROW / 8), 256, 0, stream>>>(t, ln1g, ln1b, ln16);
    gemm_f16_kernel<EPI_NONE><<<dim3(QKVN / 64, MB), 128, 0, stream>>>(
        ln16, qkvT, nullptr, nullptr, nullptr, qkv16, QKVN, DIMC);
    attn_kernel<<<2 * NWW * NWW * 6, 128, 0, stream>>>(qkv16, pos, attn16, shifted);
    gemm_f16_kernel<EPI_BIAS_RES><<<dim3(DIMC / 64, MB), 128, 0, stream>>>(
        attn16, woT, bo, t, t, nullptr, DIMC, DIMC);
    layernorm_kernel<<<(unsigned)(MROW / 8), 256, 0, stream>>>(t, ln2g, ln2b, ln16);
    gemm_f16_kernel<EPI_BIAS_GELU><<<dim3(MLPN / 64, MB), 128, 0, stream>>>(
        ln16, w1T, b1, nullptr, nullptr, h16, MLPN, DIMC);
    gemm_f16_kernel<EPI_BIAS_RES><<<dim3(DIMC / 64, MB), 128, 0, stream>>>(
        h16, w2T, b2, t, t, nullptr, DIMC, MLPN);
  };

  run_block(P[3],  P[4],  qkvT_r, P[6],  woT_r, P[8],  P[9],  P[10], w1T_r, P[12], w2T_r, P[14], 0);
  run_block(P[15], P[16], qkvT_s, P[18], woT_s, P[20], P[21], P[22], w1T_s, P[24], w2T_s, P[26], 1);

  to_nchw_kernel<<<(unsigned)((MROW * DIMC + 255) / 256), 256, 0, stream>>>(t, (float*)d_out);
}
